// DCSRM_55250459295988
// MI455X (gfx1250) — matrix operand order repaired and checked by reading, not compiled, not run on hardware
//
#include <hip/hip_runtime.h>
#include <hip/hip_bf16.h>
#include <math.h>

// SRM + dynamic depthwise conv for MI455X (gfx1250, wave32).
// Pass 1: per-(n,c) sum/sumsq reduction over 96*96 -> gate g, gap = g*mean.
// Pass 2: tiny MLP via v_wmma_f32_16x16x32_f16; folds gate + bias into the
//         per-(n,c) 3x3 kernel (conv(g*x,k) == conv(x, g*k)).
// Pass 3: LDS-tiled 3x3 depthwise conv over raw x.

#define NB   16
#define CC   256
#define HH   96
#define WW   96
#define HW   (HH * WW)          // 9216
#define NC   (NB * CC)          // 4096
#define HID  16                 // C / RED
#define KK   9                  // 3x3
#define OUTF (CC * KK)          // 2304
#define EPSV 1e-5f

typedef __attribute__((ext_vector_type(16))) _Float16 v16h;
typedef __attribute__((ext_vector_type(8)))  float    v8f;

// ---------------------------------------------------------------------------
// Kernel 1: per-(n,c) stats -> g = sigmoid(mean*cfc0 + std*cfc1), gap = g*mean
// One block (256 threads) per (n,c); float4 loads, LDS tree reduction.
// ---------------------------------------------------------------------------
__global__ void srm_stats_kernel(const float* __restrict__ x,
                                 const float* __restrict__ cfc,
                                 float* __restrict__ g,
                                 float* __restrict__ gap)
{
    const int nc  = blockIdx.x;            // 0..4095
    const int c   = nc & (CC - 1);
    const int tid = threadIdx.x;
    const float4* xv = (const float4*)(x + (size_t)nc * HW);   // 2304 float4

    float s = 0.f, ss = 0.f;
    #pragma unroll 3
    for (int i = tid; i < HW / 4; i += 256) {
        float4 v = xv[i];
        s  += (v.x + v.y) + (v.z + v.w);
        ss += (v.x * v.x + v.y * v.y) + (v.z * v.z + v.w * v.w);
    }

    __shared__ float sb[256];
    __shared__ float qb[256];
    sb[tid] = s; qb[tid] = ss;
    __syncthreads();
    for (int off = 128; off > 0; off >>= 1) {
        if (tid < off) { sb[tid] += sb[tid + off]; qb[tid] += qb[tid + off]; }
        __syncthreads();
    }
    if (tid == 0) {
        const float sum = sb[0], sumsq = qb[0];
        const float mean = sum * (1.0f / HW);
        const float var  = (sumsq - sum * mean) * (1.0f / (HW - 1)) + EPSV; // unbiased
        const float sd   = sqrtf(var);
        const float z    = mean * cfc[2 * c] + sd * cfc[2 * c + 1];
        const float gg   = 1.0f / (1.0f + expf(-z));
        g[nc]   = gg;
        gap[nc] = gg * mean;   // mean of srm = g * mean(x)
    }
}

// ---------------------------------------------------------------------------
// Kernel 2: dynamic-weight MLP on the matrix core.
//   hid  = relu(gap[16,256] @ w1^T[256,16] + b1)      (wave 0, 8x WMMA K=32)
//   dyn  = hid[16,16]       @ w2^T[16,2304] + b2      (8 waves, 144 tiles)
//   kdyn[n, c*9+j] = (dyn + b2) * g[n,c]              (gate folded in)
// Single block, 256 threads = 8 waves.
// ---------------------------------------------------------------------------
__global__ void srm_mlp_wmma_kernel(const float* __restrict__ gap,
                                    const float* __restrict__ g,
                                    const float* __restrict__ w1,
                                    const float* __restrict__ b1,
                                    const float* __restrict__ w2,
                                    const float* __restrict__ b2,
                                    float* __restrict__ kdyn)
{
    __shared__ _Float16 hidS[16 * 16];

    const int tid  = threadIdx.x;
    const int lane = tid & 31;
    const int wave = tid >> 5;
    const int m    = lane & 15;    // M row (A), N col (B/C/D)
    const int half = lane >> 4;    // lane half selects K/M striping

    // ---- GEMM1 on wave 0: hid = relu(gap @ w1^T + b1), M=N=16, K=256 ----
    if (wave == 0) {
        v8f acc = {};
        for (int k0 = 0; k0 < CC; k0 += 32) {
            v16h a, b;
            #pragma unroll
            for (int v = 0; v < 8; ++v) {
                // A (16-bit 16x32): lane m holds row M=m; VGPR v -> K pair
                const int ka = k0 + (v >> 2) * 16 + half * 8 + (v & 3) * 2;
                a[2 * v]     = (_Float16)gap[m * CC + ka];
                a[2 * v + 1] = (_Float16)gap[m * CC + ka + 1];
                // B (16-bit 32x16): lane m holds col N=m; B[k][r] = w1[r][k]
                const int kb = k0 + (v >> 2) * 16 + half * 8 + (v & 3) * 2;
                b[2 * v]     = (_Float16)w1[m * CC + kb];
                b[2 * v + 1] = (_Float16)w1[m * CC + kb + 1];
            }
            acc = __builtin_amdgcn_wmma_f32_16x16x32_f16(
                      false, a, false, b, (short)0, acc, false, false);
        }
        // C/D layout: lane m = col N, VGPR v = row M = v + half*8
        #pragma unroll
        for (int v = 0; v < 8; ++v) {
            float h = acc[v] + b1[m];
            hidS[(v + half * 8) * HID + m] = (_Float16)(h > 0.f ? h : 0.f);
        }
    }
    __syncthreads();

    // ---- GEMM2: dyn = hid @ w2^T, M=16, K=16 (padded to 32), N=2304 ----
    v16h a = {};
    #pragma unroll
    for (int v = 0; v < 8; ++v) {
        const int ka = (v >> 2) * 16 + half * 8 + (v & 3) * 2;
        if (ka < HID) {                      // K in [16,32) zero-padded
            a[2 * v]     = hidS[m * HID + ka];
            a[2 * v + 1] = hidS[m * HID + ka + 1];
        }
    }

    for (int t = wave; t < OUTF / 16; t += 8) {   // 144 tiles / 8 waves
        const int o = t * 16 + m;                 // output feature (c*9+j)
        v16h b = {};
        #pragma unroll
        for (int v = 0; v < 8; ++v) {
            const int kb = (v >> 2) * 16 + half * 8 + (v & 3) * 2;     // B[k][o] = w2[o][k]
            if (kb < HID) {
                b[2 * v]     = (_Float16)w2[o * HID + kb];
                b[2 * v + 1] = (_Float16)w2[o * HID + kb + 1];
            }
        }
        v8f acc = {};
        acc = __builtin_amdgcn_wmma_f32_16x16x32_f16(
                  false, a, false, b, (short)0, acc, false, false);
        #pragma unroll
        for (int v = 0; v < 8; ++v) {
            const int n = v + half * 8;           // batch row
            const float d = acc[v] + b2[o];
            kdyn[n * OUTF + o] = d * g[n * CC + o / KK];   // fold gate
        }
    }
}

// ---------------------------------------------------------------------------
// Kernel 3: per-(n,c) 3x3 depthwise conv over raw x with folded kernel kdyn.
// grid = (4096, 6); each block: 16 output rows x 96 cols, 18x98 LDS tile.
// ---------------------------------------------------------------------------
#define TROWS 16
#define TLDSW 98   // 96 + 2 halo cols

__global__ void srm_dwconv_kernel(const float* __restrict__ x,
                                  const float* __restrict__ kdyn,
                                  float* __restrict__ out)
{
    const int nc      = blockIdx.x;
    const int rowBase = blockIdx.y * TROWS;
    const int n = nc >> 8, c = nc & (CC - 1);
    const float* __restrict__ xp = x   + (size_t)nc * HW;
    float*       __restrict__ op = out + (size_t)nc * HW;

    // 9 block-uniform kernel coefficients (already gate-scaled + biased)
    float kc[9];
    const float* kp = kdyn + (size_t)n * OUTF + c * KK;
    #pragma unroll
    for (int j = 0; j < 9; ++j) kc[j] = kp[j];

    __shared__ float tile[(TROWS + 2) * TLDSW];   // 18*98 = 1764 floats

    const int tid = threadIdx.x;
    for (int i = tid; i < (TROWS + 2) * TLDSW; i += 256) {
        const int rr = i / TLDSW;          // 0..17 -> global row rowBase-1+rr
        const int cc = i - rr * TLDSW;     // 0..97 -> global col cc-1
        const int gr = rowBase - 1 + rr;
        const int gc = cc - 1;
        float v = 0.f;
        if ((unsigned)gr < (unsigned)HH && (unsigned)gc < (unsigned)WW)
            v = xp[gr * WW + gc];
        tile[i] = v;
    }
    __syncthreads();

    for (int p = tid; p < TROWS * WW; p += 256) {
        const int r  = p / WW;
        const int cc = p - r * WW;
        const float* t0 = &tile[r * TLDSW + cc];
        float acc = 0.f;
        #pragma unroll
        for (int dy = 0; dy < 3; ++dy) {
            #pragma unroll
            for (int dx = 0; dx < 3; ++dx)
                acc = fmaf(kc[dy * 3 + dx], t0[dy * TLDSW + dx], acc);
        }
        op[(rowBase + r) * WW + cc] = acc;
    }
}

// ---------------------------------------------------------------------------
extern "C" void kernel_launch(void* const* d_in, const int* in_sizes, int n_in,
                              void* d_out, int out_size, void* d_ws, size_t ws_size,
                              hipStream_t stream)
{
    const float* x   = (const float*)d_in[0];   // [16,256,96,96]
    const float* cfc = (const float*)d_in[1];   // [256,2]
    const float* w1  = (const float*)d_in[2];   // [16,256]
    const float* b1  = (const float*)d_in[3];   // [16]
    const float* w2  = (const float*)d_in[4];   // [2304,16]
    const float* b2  = (const float*)d_in[5];   // [2304]
    float* out = (float*)d_out;                 // [16,256,96,96]

    float* g    = (float*)d_ws;                 // [4096]
    float* gap  = g + NC;                       // [4096]
    float* kdyn = gap + NC;                     // [16*2304]

    srm_stats_kernel<<<NC, 256, 0, stream>>>(x, cfc, g, gap);
    srm_mlp_wmma_kernel<<<1, 256, 0, stream>>>(gap, g, w1, b1, w2, b2, kdyn);
    dim3 grid3(NC, HH / TROWS);
    srm_dwconv_kernel<<<grid3, 256, 0, stream>>>(x, kdyn, out);
}
